// CombinedGeomAttention_67697274520078
// MI455X (gfx1250) — matrix operand order repaired and checked by reading, not compiled, not run on hardware
//
#include <hip/hip_runtime.h>

// ---------------- problem constants (fixed by reference) ----------------
#define NB 4
#define NH 8
#define NL 2048
#define NS 2048
#define NE 64
#define BM 128              // query rows per block (8 waves x 16)
#define BN 128              // key cols per S-iteration
#define NWAVE 8
#define ROWSTR (NH*NE)      // 512 floats between consecutive l (or s) rows
#define KS_STR 72           // bf16 elems per K-tile row in LDS (64 + 8 pad)
#define VT_STR 136          // bf16 elems per V^T row in LDS (128 + 8 pad)
#define P_STR  72           // bf16 elems per P-half row in LDS (64 + 8 pad)
#define NITER (NS / BN)     // 16

typedef __bf16 v16bf __attribute__((ext_vector_type(16)));
typedef float  v8f   __attribute__((ext_vector_type(8)));
typedef unsigned int v4u __attribute__((ext_vector_type(4)));
typedef float  v4f   __attribute__((ext_vector_type(4)));

union BF16Frag {
  v16bf bf;
  v4u   q[2];
  unsigned short u[16];
};

__device__ __forceinline__ unsigned short f2bf(float f) {
  unsigned int u = __float_as_uint(f);
  u += 0x7FFFu + ((u >> 16) & 1u);   // round-to-nearest-even
  return (unsigned short)(u >> 16);
}

struct __align__(16) Smem {
  unsigned short ks[BN * KS_STR];        // K tile bf16, [s][e]      18432 B
  unsigned short vt[NE * VT_STR];        // V^T tile bf16, [e][s]    17408 B
  unsigned short p[NWAVE][16 * P_STR];   // per-wave P half [m][s64] 18432 B
  float kn2[BN];
  float qn2[BM];
  float red[NWAVE];
};                                        // total ~55.3 KB

__global__ void zero_mean_kernel(float* p) { *p = 0.0f; }

__global__ __launch_bounds__(256, 1)
void geom_attn_kernel(const float* __restrict__ Qg, const float* __restrict__ Kg,
                      const float* __restrict__ Vg, float* __restrict__ Og,
                      float* __restrict__ meanAcc)
{
  __shared__ Smem sm;

  const int tid   = threadIdx.x;
  const int wave  = tid >> 5;
  const int lane  = tid & 31;
  const int lhalf = lane >> 4;       // 0 or 1
  const int l16   = lane & 15;
  const int kb    = lhalf * 8;       // A-layout K quarter base

  const int blk = blockIdx.x;
  const int mt  = blk & (NL/BM - 1); // 0..15
  const int bh  = blk >> 4;
  const int h   = bh & (NH - 1);
  const int b   = bh >> 3;

  const float scale = 0.125f;        // 1/sqrt(64)

  const float* Qb = Qg + (size_t)b * NL * ROWSTR + (size_t)h * NE;
  const float* Kb = Kg + (size_t)b * NS * ROWSTR + (size_t)h * NE;
  const float* Vb = Vg + (size_t)b * NS * ROWSTR + (size_t)h * NE;
  float*       Ob = Og + (size_t)b * NL * ROWSTR + (size_t)h * NE;

  // ---- |q|^2 for the 128 rows of this block (2 threads per row) ----
  {
    const int row = tid >> 1;
    const int e0  = (tid & 1) * 32;
    const float* qr = Qb + (size_t)(mt * BM + row) * ROWSTR + e0;
    float s = 0.0f;
#pragma unroll
    for (int i = 0; i < 8; ++i) {
      v4f f = ((const v4f*)qr)[i];
      s += f.x*f.x + f.y*f.y + f.z*f.z + f.w*f.w;
    }
    s += __shfl_xor(s, 1, 32);
    if ((tid & 1) == 0) sm.qn2[row] = s;
  }

  // ---- load Q A-fragments (16 rows per wave), fp32 -> bf16, A layout ----
  // lane holds row m = l16; elements 0..7 <- K [kbase, kbase+8),
  // elements 8..15 <- K [kbase+16, kbase+24), kbase = kstep*32 + lhalf*8
  BF16Frag qa[2];
  {
    const int grow = mt * BM + wave * 16 + l16;
    const float* qrow = Qb + (size_t)grow * ROWSTR;
#pragma unroll
    for (int ks = 0; ks < 2; ++ks) {
      const int e0 = ks * 32 + kb;
#pragma unroll
      for (int c = 0; c < 2; ++c) {
        v4f f0 = *(const v4f*)(qrow + e0 + c * 16);
        v4f f1 = *(const v4f*)(qrow + e0 + c * 16 + 4);
        const int base = c * 8;
        qa[ks].u[base+0] = f2bf(f0.x); qa[ks].u[base+1] = f2bf(f0.y);
        qa[ks].u[base+2] = f2bf(f0.z); qa[ks].u[base+3] = f2bf(f0.w);
        qa[ks].u[base+4] = f2bf(f1.x); qa[ks].u[base+5] = f2bf(f1.y);
        qa[ks].u[base+6] = f2bf(f1.z); qa[ks].u[base+7] = f2bf(f1.w);
      }
    }
  }

  __syncthreads();

  // per-lane |q|^2 for the rows this lane's C-layout registers cover
  float qn2r[8];
#pragma unroll
  for (int r = 0; r < 8; ++r) qn2r[r] = sm.qn2[wave * 16 + r + 8 * lhalf];

  v8f o[4];
#pragma unroll
  for (int t = 0; t < 4; ++t)
#pragma unroll
    for (int r = 0; r < 8; ++r) o[t][r] = 0.0f;

  float mrow[8], lrow[8];
#pragma unroll
  for (int r = 0; r < 8; ++r) { mrow[r] = -INFINITY; lrow[r] = 0.0f; }
  float sumabs = 0.0f;

  // =================== sweep over S in 128-column tiles ===================
  for (int it = 0; it < NITER; ++it) {
    const int sb = it * BN;
    __syncthreads();   // previous tile's LDS consumers done before restage

    // ---- stage K tile -> LDS bf16 [s][e], compute |k|^2 ----
    {
      const int row = tid >> 1;
      const int e0  = (tid & 1) * 32;
      const float* kr = Kb + (size_t)(sb + row) * ROWSTR + e0;
      if (it + 1 < NITER) __builtin_prefetch(kr + (size_t)BN * ROWSTR, 0, 1);
      float acc2 = 0.0f;
#pragma unroll
      for (int i = 0; i < 4; ++i) {
        v4f a = ((const v4f*)kr)[2*i];
        v4f c = ((const v4f*)kr)[2*i+1];
        acc2 += a.x*a.x + a.y*a.y + a.z*a.z + a.w*a.w
              + c.x*c.x + c.y*c.y + c.z*c.z + c.w*c.w;
        v4u u;
        u.x = (unsigned)f2bf(a.x) | ((unsigned)f2bf(a.y) << 16);
        u.y = (unsigned)f2bf(a.z) | ((unsigned)f2bf(a.w) << 16);
        u.z = (unsigned)f2bf(c.x) | ((unsigned)f2bf(c.y) << 16);
        u.w = (unsigned)f2bf(c.z) | ((unsigned)f2bf(c.w) << 16);
        *(v4u*)&sm.ks[row * KS_STR + e0 + i * 8] = u;
      }
      acc2 += __shfl_xor(acc2, 1, 32);
      if ((tid & 1) == 0) sm.kn2[row] = acc2;
    }

    // ---- stage V tile transposed -> LDS bf16 [e][s] ----
    {
      const int row = tid >> 1;
      const int e0  = (tid & 1) * 32;
      const float* vr = Vb + (size_t)(sb + row) * ROWSTR + e0;
      if (it + 1 < NITER) __builtin_prefetch(vr + (size_t)BN * ROWSTR, 0, 1);
#pragma unroll
      for (int i = 0; i < 8; ++i) {
        v4f f = ((const v4f*)vr)[i];
        const int e = e0 + i * 4;
        sm.vt[(e+0) * VT_STR + row] = f2bf(f.x);
        sm.vt[(e+1) * VT_STR + row] = f2bf(f.y);
        sm.vt[(e+2) * VT_STR + row] = f2bf(f.z);
        sm.vt[(e+3) * VT_STR + row] = f2bf(f.w);
      }
    }
    __syncthreads();

    // ---- scores: S = Q K^T via bf16 WMMA, then wedge transform ----
    v8f sc[8];
#pragma unroll
    for (int j = 0; j < 8; ++j) {
      v8f acc;
#pragma unroll
      for (int r = 0; r < 8; ++r) acc[r] = 0.0f;
#pragma unroll
      for (int ks = 0; ks < 2; ++ks) {
        BF16Frag bfr;   // B layout: lane = col s, e [8h,8h+8) and [16+8h,24+8h)
        const v4u* src = (const v4u*)&sm.ks[(j*16 + l16) * KS_STR + lhalf*8 + ks*32];
        bfr.q[0] = src[0];
        bfr.q[1] = src[2];
        acc = __builtin_amdgcn_wmma_f32_16x16x32_bf16(
            false, qa[ks].bf, false, bfr.bf, (short)0, acc, false, false);
      }
      const float kn2j = sm.kn2[j*16 + l16];
#pragma unroll
      for (int r = 0; r < 8; ++r) {
        float d  = acc[r];
        float w2 = fmaxf(qn2r[r] * kn2j - d * d, 0.0f);
        float s  = sqrtf(w2 + 1e-8f) * scale;   // ALPHA == 1
        sumabs += s;                            // scores are >= 0
        sc[j][r] = s;
      }
    }

    // ---- online softmax update (rows live across 16-lane halves) ----
    float corr[8];
#pragma unroll
    for (int r = 0; r < 8; ++r) {
      float tm = sc[0][r];
#pragma unroll
      for (int j = 1; j < 8; ++j) tm = fmaxf(tm, sc[j][r]);
      tm = fmaxf(tm, __shfl_xor(tm, 1, 32));
      tm = fmaxf(tm, __shfl_xor(tm, 2, 32));
      tm = fmaxf(tm, __shfl_xor(tm, 4, 32));
      tm = fmaxf(tm, __shfl_xor(tm, 8, 32));
      const float mn = fmaxf(mrow[r], tm);
      const float c  = __expf(mrow[r] - mn);
      mrow[r] = mn;
      float rs = 0.0f;
#pragma unroll
      for (int j = 0; j < 8; ++j) {
        float pv = __expf(sc[j][r] - mn);
        sc[j][r] = pv;
        rs += pv;
      }
      rs += __shfl_xor(rs, 1, 32);
      rs += __shfl_xor(rs, 2, 32);
      rs += __shfl_xor(rs, 4, 32);
      rs += __shfl_xor(rs, 8, 32);
      lrow[r] = lrow[r] * c + rs;
      corr[r] = c;
    }
#pragma unroll
    for (int t = 0; t < 4; ++t)
#pragma unroll
      for (int r = 0; r < 8; ++r) o[t][r] *= corr[r];

    // ---- A*V: C->A relayout of P through per-wave LDS, two 64-col halves ----
    unsigned short* pw = &sm.p[wave][0];
#pragma unroll
    for (int half = 0; half < 2; ++half) {
#pragma unroll
      for (int jj = 0; jj < 4; ++jj) {
        const int j = half * 4 + jj;
#pragma unroll
        for (int r = 0; r < 8; ++r) {
          pw[(r + 8*lhalf) * P_STR + jj*16 + l16] = f2bf(sc[j][r]);
        }
      }
      asm volatile("s_wait_dscnt 0x0" ::: "memory");   // wave-local LDS RAW
#pragma unroll
      for (int ks2 = 0; ks2 < 2; ++ks2) {
        BF16Frag afr;   // A layout over K = s-local in [0,64)
        afr.q[0] = *(const v4u*)&pw[l16 * P_STR + ks2*32 + kb];
        afr.q[1] = *(const v4u*)&pw[l16 * P_STR + ks2*32 + kb + 16];
#pragma unroll
        for (int t = 0; t < 4; ++t) {
          BF16Frag bfr; // B layout: lane = col e, s [8h,8h+8) and [16+8h,24+8h)
          const v4u* pv = (const v4u*)
              &sm.vt[(t*16 + l16) * VT_STR + half*64 + ks2*32 + lhalf*8];
          bfr.q[0] = pv[0];
          bfr.q[1] = pv[2];
          o[t] = __builtin_amdgcn_wmma_f32_16x16x32_bf16(
              false, afr.bf, false, bfr.bf, (short)0, o[t], false, false);
        }
      }
      asm volatile("s_wait_dscnt 0x0" ::: "memory");   // before next half's stores
    }
  }

  // ---- normalize and write output (C layout -> [l][h][e]) ----
#pragma unroll
  for (int r = 0; r < 8; ++r) {
    const float inv = 1.0f / lrow[r];
    const int grow = mt * BM + wave * 16 + r + 8 * lhalf;
    float* orow = Ob + (size_t)grow * ROWSTR;
#pragma unroll
    for (int t = 0; t < 4; ++t)
      orow[t*16 + l16] = o[t][r] * inv;
  }

  // ---- block reduce sum(|scores|), one atomic per block ----
  sumabs += __shfl_xor(sumabs, 16, 32);
  sumabs += __shfl_xor(sumabs,  8, 32);
  sumabs += __shfl_xor(sumabs,  4, 32);
  sumabs += __shfl_xor(sumabs,  2, 32);
  sumabs += __shfl_xor(sumabs,  1, 32);
  if (lane == 0) sm.red[wave] = sumabs;
  __syncthreads();
  if (tid == 0) {
    float s = 0.0f;
#pragma unroll
    for (int w = 0; w < NWAVE; ++w) s += sm.red[w];
    atomicAdd(meanAcc, s * (1.0f / ((float)NB * NH * (float)NL * (float)NS)));
  }
}

extern "C" void kernel_launch(void* const* d_in, const int* in_sizes, int n_in,
                              void* d_out, int out_size, void* d_ws, size_t ws_size,
                              hipStream_t stream) {
  const float* Q = (const float*)d_in[0];
  const float* K = (const float*)d_in[1];
  const float* V = (const float*)d_in[2];
  float* out = (float*)d_out;
  float* mean_out = out + (size_t)NB * NL * NH * NE;

  hipLaunchKernelGGL(zero_mean_kernel, dim3(1), dim3(1), 0, stream, mean_out);
  hipLaunchKernelGGL(geom_attn_kernel, dim3(NB * NH * (NL / BM)), dim3(256), 0,
                     stream, Q, K, V, out, mean_out);
}
